// PFNet_56873956934073
// MI455X (gfx1250) — matrix operand order repaired and checked by reading, not compiled, not run on hardware
//
#include <hip/hip_runtime.h>
#include <math.h>

// ---------------------------------------------------------------------------
// PFNet forward for MI455X (gfx1250, wave32, WMMA, bf16 operands / f32 accum).
// All GEMM operands live in memory as bf16 laid out so each WMMA operand is
// fetched with contiguous b128 vector loads (A: two 8-elem K-runs per lane,
// B: two 8-elem K-runs per lane, via transposed weight/Hc copies).
// The dominant adj @ Hc GEMM computes the 16x32 adj tile on the fly into a
// double-buffered LDS tile (1 barrier per K-step) using native v_exp/v_rsq,
// and starts its K loop at the diagonal tile (triu -> ~half the work).
// ---------------------------------------------------------------------------

#define NROWS 12000
#define LDB   288        // padded activation stride (multiple of 32)

typedef __attribute__((ext_vector_type(16))) __bf16        v16bf;
typedef __attribute__((ext_vector_type(8)))  float         v8f;
typedef __attribute__((ext_vector_type(4)))  unsigned int  uint4v;

union Op16 { v16bf v; uint4v q[2]; unsigned short s[16]; };

#define LOG2E 1.4426950408889634f

__device__ __forceinline__ unsigned short f32_to_bf16(float f) {
  unsigned int u = __float_as_uint(f);
  u += 0x7FFFu + ((u >> 16) & 1u);     // round-to-nearest-even
  return (unsigned short)(u >> 16);
}

// native v_exp_f32-based exp (plenty for bf16-bound results)
__device__ __forceinline__ float fast_exp(float x) {
  return __builtin_amdgcn_exp2f(x * LOG2E);
}

__device__ __forceinline__ float selu_f(float x) {
  // scale = 1.0507009873554805, scale*alpha = 1.7580993408473766
  return x > 0.f ? 1.0507009873554805f * x
                 : 1.7580993408473766f * (fast_exp(x) - 1.f);
}

// ---------------------------------------------------------------------------
// Fused one-hot encode + input1 dense + selu -> bf16 activations.
// ---------------------------------------------------------------------------
__global__ void encode_input1(const float* __restrict__ X,
                              const float* __restrict__ W1,   // 18 x 256
                              const float* __restrict__ b1,   // 256
                              unsigned short* __restrict__ out) { // N x LDB bf16
  int idx = blockIdx.x * blockDim.x + threadIdx.x;
  if (idx >= NROWS * 256) return;
  int row = idx >> 8, c = idx & 255;
  const float* xr = X + row * 8;
  int id = (int)xr[0];
  id = id < 0 ? 0 : (id > 10 ? 10 : id);
  float acc = b1[c] + W1[id * 256 + c];
#pragma unroll
  for (int p = 0; p < 7; ++p) acc += xr[1 + p] * W1[(11 + p) * 256 + c];
  out[(size_t)row * LDB + c] = f32_to_bf16(selu_f(acc));
}

// ---------------------------------------------------------------------------
// Dense layer: Y = act(Xb @ WT^T + b).  No LDS, no barriers: pure b128 loads
// + v_wmma_f32_16x16x32_bf16.  Block = 128 (4 waves), tile 16 rows x 64 cols.
// act: 0=none, 1=selu, 2=tanh.  Yb (bf16) / Yf (f32) may each be null.
// ---------------------------------------------------------------------------
__global__ __launch_bounds__(128) void dense_wmma(
    const unsigned short* __restrict__ Xb, int ldx,
    const unsigned short* __restrict__ WT, int dinP,   // WT: [doutP][dinP] bf16
    const float* __restrict__ bias,                    // doutP f32 (padded)
    unsigned short* __restrict__ Yb, float* __restrict__ Yf, int ldy,
    int doutP, int act) {
  const int lane = threadIdx.x & 31;
  const int wave = threadIdx.x >> 5;
  const int g = lane >> 4, ln = lane & 15;
  const int r0 = blockIdx.x * 16;
  const int wc = blockIdx.y * 64 + wave * 16;
  if (wc >= doutP) return;

  const unsigned short* arow = Xb + (size_t)(r0 + ln) * ldx;
  const unsigned short* brow = WT + (size_t)(wc + ln) * dinP;
  v8f acc = {};
  for (int k = 0; k < dinP; k += 32) {
    Op16 a, b;
    // A layout: e<8 -> k = 8g+e ; e>=8 -> k = 16+8g+(e-8)  (two b128 runs)
    a.q[0] = *(const uint4v*)(arow + k + 8 * g);
    a.q[1] = *(const uint4v*)(arow + k + 16 + 8 * g);
    // B layout: e<8 -> k = 8g+e ; e>=8 -> k = 16+8g+(e-8), n = lane&15  (two b128 runs)
    b.q[0] = *(const uint4v*)(brow + k + 8 * g);
    b.q[1] = *(const uint4v*)(brow + k + 16 + 8 * g);
    acc = __builtin_amdgcn_wmma_f32_16x16x32_bf16(false, a.v, false, b.v,
                                                  (short)0, acc, false, false);
  }
  const float bv = bias[wc + ln];
#pragma unroll
  for (int r = 0; r < 8; ++r) {
    int m = r + (g << 3);            // C/D layout: m = vgpr + laneGroup*8
    float y = acc[r] + bv;
    if (act == 1)      y = selu_f(y);
    else if (act == 2) y = tanhf(y);
    if (Yb) Yb[(size_t)(r0 + m) * ldy + wc + ln] = f32_to_bf16(y);
    if (Yf) Yf[(size_t)(r0 + m) * ldy + wc + ln] = y;
  }
}

// ---------------------------------------------------------------------------
// Adjacency GEMM: Xout = selu( triu(adj) @ Hc ), adj computed on the fly.
// Block = 256 threads (8 waves): 16 rows x 256 cols.  Double-buffered adj
// tile in LDS (bf16, one barrier per K-step); Hc read from the transposed
// bf16 copy (L2-resident) with b128 loads; native v_rsq/v_exp/v_rcp sigmoid.
// ---------------------------------------------------------------------------
__global__ __launch_bounds__(256) void adj_wmma(
    const float* __restrict__ hF,            // N x LDB f32 (cols 0..2 = pos)
    const unsigned short* __restrict__ HcT,  // 256 x NROWS bf16 (transposed)
    unsigned short* __restrict__ Xout,       // N x LDB bf16
    const float* __restrict__ pa, const float* __restrict__ pb) {
  __shared__ __align__(16) unsigned short As[2][16][40]; // 80B rows, 16B align
  const float dist_a = pa[0], dist_b = pb[0];
  const int t = threadIdx.x;
  const int wave = t >> 5, lane = t & 31;
  const int g = lane >> 4, ln = lane & 15;
  const int i0 = blockIdx.x * 16;
  const int c0 = wave * 32;                  // wave covers cols c0..c0+31
  const int r  = t >> 5;                     // this thread owns rows r, r+8
  const int jj = t & 31;

  const float aix0 = hF[(size_t)(i0 + r) * LDB + 0];
  const float aiy0 = hF[(size_t)(i0 + r) * LDB + 1];
  const float aiz0 = hF[(size_t)(i0 + r) * LDB + 2];
  const float aix1 = hF[(size_t)(i0 + r + 8) * LDB + 0];
  const float aiy1 = hF[(size_t)(i0 + r + 8) * LDB + 1];
  const float aiz1 = hF[(size_t)(i0 + r + 8) * LDB + 2];

  const unsigned short* brow0 = HcT + (size_t)(c0 + ln) * NROWS;
  const unsigned short* brow1 = HcT + (size_t)(c0 + 16 + ln) * NROWS;

  v8f acc0 = {}, acc1 = {};
  int p = 0;
  for (int jb = (i0 >> 5) << 5; jb < NROWS; jb += 32, p ^= 1) {
    const int j = jb + jj;
    const float ajx = hF[(size_t)j * LDB + 0];
    const float ajy = hF[(size_t)j * LDB + 1];
    const float ajz = hF[(size_t)j * LDB + 2];
    {
      float dx = aix0 - ajx, dy = aiy0 - ajy, dz = aiz0 - ajz;
      float d2 = dx * dx + dy * dy + dz * dz;
      float dinv = d2 > 0.f ? __builtin_amdgcn_rsqf(d2) : 0.f;
      float e = __builtin_amdgcn_exp2f(-LOG2E * (dinv + dist_b));
      float sg = __builtin_amdgcn_rcpf(1.f + e);
      As[p][r][jj] = f32_to_bf16((j >= i0 + r) ? dist_a * sg : 0.f);
    }
    {
      float dx = aix1 - ajx, dy = aiy1 - ajy, dz = aiz1 - ajz;
      float d2 = dx * dx + dy * dy + dz * dz;
      float dinv = d2 > 0.f ? __builtin_amdgcn_rsqf(d2) : 0.f;
      float e = __builtin_amdgcn_exp2f(-LOG2E * (dinv + dist_b));
      float sg = __builtin_amdgcn_rcpf(1.f + e);
      As[p][r + 8][jj] = f32_to_bf16((j >= i0 + r + 8) ? dist_a * sg : 0.f);
    }
    __syncthreads();   // single barrier: next iter writes the other buffer
    Op16 a, b0, b1;
    a.q[0] = *(const uint4v*)&As[p][ln][8 * g];
    a.q[1] = *(const uint4v*)&As[p][ln][16 + 8 * g];
    b0.q[0] = *(const uint4v*)(brow0 + jb + 8 * g);
    b0.q[1] = *(const uint4v*)(brow0 + jb + 16 + 8 * g);
    b1.q[0] = *(const uint4v*)(brow1 + jb + 8 * g);
    b1.q[1] = *(const uint4v*)(brow1 + jb + 16 + 8 * g);
    acc0 = __builtin_amdgcn_wmma_f32_16x16x32_bf16(false, a.v, false, b0.v,
                                                   (short)0, acc0, false, false);
    acc1 = __builtin_amdgcn_wmma_f32_16x16x32_bf16(false, a.v, false, b1.v,
                                                   (short)0, acc1, false, false);
  }
#pragma unroll
  for (int rr = 0; rr < 8; ++rr) {
    int m = rr + (g << 3);
    Xout[(size_t)(i0 + m) * LDB + c0 + ln]      = f32_to_bf16(selu_f(acc0[rr]));
    Xout[(size_t)(i0 + m) * LDB + c0 + 16 + ln] = f32_to_bf16(selu_f(acc1[rr]));
  }
}

// ---------------------------------------------------------------------------
// Tiled transpose: Hc (N x LDB bf16, cols 0..255) -> HcT (256 x NROWS bf16)
// ---------------------------------------------------------------------------
__global__ __launch_bounds__(256) void transpose_hc(
    const unsigned short* __restrict__ src,
    unsigned short* __restrict__ dst) {
  __shared__ unsigned short tile[32][33];
  const int j0 = blockIdx.x * 32, c0 = blockIdx.y * 32;
  const int t = threadIdx.x;
  for (int p = t; p < 1024; p += 256) {
    int rr = p >> 5, cc = p & 31;
    tile[rr][cc] = src[(size_t)(j0 + rr) * LDB + c0 + cc];
  }
  __syncthreads();
  for (int p = t; p < 1024; p += 256) {
    int rr = p >> 5, cc = p & 31;
    dst[(size_t)(c0 + rr) * NROWS + j0 + cc] = tile[cc][rr];
  }
}

// ---------------------------------------------------------------------------
// Weight pack: W f32 [din x dout] -> WT bf16 [doutP x dinP], zero padded.
// ---------------------------------------------------------------------------
__global__ void pack_wT(const float* __restrict__ W, int din, int dout,
                        unsigned short* __restrict__ WT, int dinP, int doutP) {
  int idx = blockIdx.x * blockDim.x + threadIdx.x;
  if (idx >= dinP * doutP) return;
  int c = idx / dinP, k = idx % dinP;
  float v = (c < dout && k < din) ? W[k * dout + c] : 0.f;
  WT[idx] = f32_to_bf16(v);
}

// f32 zero-pad packer (biases)
__global__ void pack_pad(const float* __restrict__ src, int n,
                         float* __restrict__ dst, int np) {
  int idx = blockIdx.x * blockDim.x + threadIdx.x;
  if (idx >= np) return;
  dst[idx] = (idx < n) ? src[idx] : 0.f;
}

// xm = [x (256, bf16 copy) | selu(logits) (11) | zeros -> 288]
__global__ void build_xm(const unsigned short* __restrict__ xb,
                         const float* __restrict__ logits,
                         unsigned short* __restrict__ xm) {
  int idx = blockIdx.x * blockDim.x + threadIdx.x;
  if (idx >= NROWS * LDB) return;
  int row = idx / LDB, c = idx % LDB;
  unsigned short v;
  if (c < 256)      v = xb[(size_t)row * LDB + c];
  else if (c < 267) v = f32_to_bf16(selu_f(logits[row * 16 + (c - 256)]));
  else              v = 0;
  xm[idx] = v;
}

__global__ void final_assemble(const float* __restrict__ X,
                               const float* __restrict__ logits,
                               const float* __restrict__ pc,
                               float* __restrict__ out) {
  int row = blockIdx.x * blockDim.x + threadIdx.x;
  if (row >= NROWS) return;
  const float* xr = X + row * 8;
  float* o = out + row * 14;
#pragma unroll
  for (int c = 0; c < 11; ++c) o[c] = logits[row * 16 + c];
  float p0 = pc[row * 16 + 0], p1 = pc[row * 16 + 1], p2 = pc[row * 16 + 2];
  float eta = xr[2], phi = xr[3], E = xr[4];
  o[11] = eta + eta * p0;
  float np = phi + phi * p1;
  o[12] = atan2f(sinf(np), cosf(np));
  o[13] = E + p2;
}

// ---------------------------------------------------------------------------
// Input index map (JAX pytree leaf order: dict keys sorted):
//  0:X  1:convW 2:convB 3:dist_a 4:dist_b 5:id1W 6:id1b 7:id2W 8:id2b
//  9:id3W 10:id3b 11:in1W 12:in1b 13:in2W 14:in2b 15:in3W 16:in3b
//  17:m1W(267x267) 18:m1b 19:m2W(267x256) 20:m2b 21:m3W 22:m3b
//  23:oidW(256x11) 24:oidb 25:omW(256x3) 26:omb
// ---------------------------------------------------------------------------
extern "C" void kernel_launch(void* const* d_in, const int* in_sizes, int n_in,
                              void* d_out, int out_size, void* d_ws, size_t ws_size,
                              hipStream_t stream) {
  (void)in_sizes; (void)n_in; (void)out_size; (void)ws_size;
  const float* X     = (const float*)d_in[0];
  const float* convW = (const float*)d_in[1];
  const float* convB = (const float*)d_in[2];
  const float* dA    = (const float*)d_in[3];
  const float* dB    = (const float*)d_in[4];
  const float* id1W  = (const float*)d_in[5];
  const float* id1b  = (const float*)d_in[6];
  const float* id2W  = (const float*)d_in[7];
  const float* id2b  = (const float*)d_in[8];
  const float* id3W  = (const float*)d_in[9];
  const float* id3b  = (const float*)d_in[10];
  const float* in1W  = (const float*)d_in[11];
  const float* in1b  = (const float*)d_in[12];
  const float* in2W  = (const float*)d_in[13];
  const float* in2b  = (const float*)d_in[14];
  const float* in3W  = (const float*)d_in[15];
  const float* in3b  = (const float*)d_in[16];
  const float* m1W   = (const float*)d_in[17];
  const float* m1b   = (const float*)d_in[18];
  const float* m2W   = (const float*)d_in[19];
  const float* m2b   = (const float*)d_in[20];
  const float* m3W   = (const float*)d_in[21];
  const float* m3b   = (const float*)d_in[22];
  const float* oidW  = (const float*)d_in[23];
  const float* oidb  = (const float*)d_in[24];
  const float* omW   = (const float*)d_in[25];
  const float* omb   = (const float*)d_in[26];
  float* out = (float*)d_out;

  // ---- workspace carve (~45 MB) ----
  char* wp = (char*)d_ws;
  auto alloc = [&](size_t bytes) -> char* {
    char* r = wp; wp += (bytes + 255) & ~(size_t)255; return r;
  };
  unsigned short* b0   = (unsigned short*)alloc((size_t)NROWS * LDB * 2);
  unsigned short* b1   = (unsigned short*)alloc((size_t)NROWS * LDB * 2);
  unsigned short* b2   = (unsigned short*)alloc((size_t)NROWS * LDB * 2);
  float*          hF   = (float*)alloc((size_t)NROWS * LDB * 4);
  unsigned short* HcT  = (unsigned short*)alloc((size_t)256 * NROWS * 2);
  float*          logits = (float*)alloc((size_t)NROWS * 16 * 4);
  float*          pc   = (float*)alloc((size_t)NROWS * 16 * 4);
  unsigned short* in2T = (unsigned short*)alloc(256 * 256 * 2);
  unsigned short* in3T = (unsigned short*)alloc(256 * 256 * 2);
  unsigned short* convT= (unsigned short*)alloc(256 * 256 * 2);
  unsigned short* id1T = (unsigned short*)alloc(256 * 256 * 2);
  unsigned short* id2T = (unsigned short*)alloc(256 * 256 * 2);
  unsigned short* id3T = (unsigned short*)alloc(256 * 256 * 2);
  unsigned short* m3T  = (unsigned short*)alloc(256 * 256 * 2);
  unsigned short* oidT = (unsigned short*)alloc(16 * 256 * 2);
  unsigned short* m1T  = (unsigned short*)alloc(288 * 288 * 2);
  unsigned short* m2T  = (unsigned short*)alloc(256 * 288 * 2);
  unsigned short* omT  = (unsigned short*)alloc(16 * 256 * 2);
  float* boidp = (float*)alloc(16 * 4);
  float* bomp  = (float*)alloc(16 * 4);
  float* bm1p  = (float*)alloc(288 * 4);

  auto up = [](int n) { return (n + 255) / 256; };

  // ---- pack transposed bf16 weights + padded biases ----
  pack_wT<<<up(256 * 256), 256, 0, stream>>>(in2W, 256, 256, in2T, 256, 256);
  pack_wT<<<up(256 * 256), 256, 0, stream>>>(in3W, 256, 256, in3T, 256, 256);
  pack_wT<<<up(256 * 256), 256, 0, stream>>>(convW, 256, 256, convT, 256, 256);
  pack_wT<<<up(256 * 256), 256, 0, stream>>>(id1W, 256, 256, id1T, 256, 256);
  pack_wT<<<up(256 * 256), 256, 0, stream>>>(id2W, 256, 256, id2T, 256, 256);
  pack_wT<<<up(256 * 256), 256, 0, stream>>>(id3W, 256, 256, id3T, 256, 256);
  pack_wT<<<up(256 * 256), 256, 0, stream>>>(m3W, 256, 256, m3T, 256, 256);
  pack_wT<<<up(16 * 256), 256, 0, stream>>>(oidW, 256, 11, oidT, 256, 16);
  pack_wT<<<up(288 * 288), 256, 0, stream>>>(m1W, 267, 267, m1T, 288, 288);
  pack_wT<<<up(256 * 288), 256, 0, stream>>>(m2W, 267, 256, m2T, 288, 256);
  pack_wT<<<up(16 * 256), 256, 0, stream>>>(omW, 256, 3, omT, 256, 16);
  pack_pad<<<1, 256, 0, stream>>>(oidb, 11, boidp, 16);
  pack_pad<<<1, 256, 0, stream>>>(omb, 3, bomp, 16);
  pack_pad<<<2, 256, 0, stream>>>(m1b, 267, bm1p, 288);

  auto gemm = [&](const unsigned short* Xp, const unsigned short* WTp, int dinP,
                  const float* bp, unsigned short* Ybp, float* Yfp, int ldy,
                  int doutP, int act) {
    dim3 grid(NROWS / 16, (doutP + 63) / 64);
    dense_wmma<<<grid, 128, 0, stream>>>(Xp, LDB, WTp, dinP, bp, Ybp, Yfp, ldy,
                                         doutP, act);
  };

  // ---- encoder MLP ----
  encode_input1<<<up(NROWS * 256), 256, 0, stream>>>(X, in1W, in1b, b0);
  gemm(b0, in2T, 256, in2b, b1, nullptr, LDB, 256, 1);      // input2
  gemm(b1, in3T, 256, in3b, b2, hF, LDB, 256, 1);           // input3 -> h
  gemm(b2, convT, 256, convB, b0, nullptr, LDB, 256, 0);    // Hc (pre-act)
  transpose_hc<<<dim3(NROWS / 32, 8), 256, 0, stream>>>(b0, HcT);

  // ---- x = selu(triu(adj) @ Hc) ----
  adj_wmma<<<NROWS / 16, 256, 0, stream>>>(hF, HcT, b1, dA, dB);

  // ---- id head ----
  gemm(b1, id1T, 256, id1b, b0, nullptr, LDB, 256, 1);
  gemm(b0, id2T, 256, id2b, b2, nullptr, LDB, 256, 1);
  gemm(b2, id3T, 256, id3b, b0, nullptr, LDB, 256, 1);
  gemm(b0, oidT, 256, boidp, nullptr, logits, 16, 16, 0);   // logits (f32)

  // ---- momentum head ----
  build_xm<<<up(NROWS * LDB), 256, 0, stream>>>(b1, logits, b2);  // xm
  gemm(b2, m1T, 288, bm1p, b0, nullptr, LDB, 288, 1);
  gemm(b0, m2T, 288, m2b, b1, nullptr, LDB, 256, 1);
  gemm(b1, m3T, 256, m3b, b2, nullptr, LDB, 256, 1);
  gemm(b2, omT, 256, bomp, nullptr, pc, 16, 16, 2);         // tanh -> pred_corr

  final_assemble<<<up(NROWS), 256, 0, stream>>>(X, logits, pc, out);
}
